// MultiHeadLatentAttention_56659208568833
// MI455X (gfx1250) — matrix operand order repaired and checked by reading, not compiled, not run on hardware
//
#include <hip/hip_runtime.h>

// Problem constants (fixed by the reference)
#define HD   16      // heads
#define EE   2048    // embed
#define LATD 512     // latent
#define DH   128     // head dim
#define BB   2
#define SS   2048
#define TT   (BB*SS) // 4096 tokens

typedef __attribute__((ext_vector_type(16))) __bf16 v16bf;
typedef __attribute__((ext_vector_type(8)))  float  v8f;

union FragAB { v16bf v; unsigned int u[8]; };
union FragC  { v8f  v; float f[8]; };

// K-offset of packed pair g (0..7) for the 16-bit A-matrix layout:
// VGPR g<4 covers K 0..15 (half 0: K 0..7, half 1: K 8..15), g>=4 covers K 16..31.
__device__ __forceinline__ int koffA(int g, int half) {
  return ((g >> 2) << 4) + ((g & 3) << 1) + (half << 3);
}

// Async 16-byte global->LDS copy (per lane). Tracked by ASYNCcnt.
// VDST VGPR = LDS byte offset (flat LDS addr low 32 bits), VADDR = 64-bit global addr.
__device__ __forceinline__ void async_copy16(const __bf16* gp, __bf16* lp) {
  unsigned int loff = (unsigned int)(unsigned long long)lp; // addr[31:0] = LDS offset
  unsigned long long ga = (unsigned long long)gp;
  asm volatile("global_load_async_to_lds_b128 %0, %1, off"
               :: "v"(loff), "v"(ga) : "memory");
}
__device__ __forceinline__ void wait_async0() {
  asm volatile("s_wait_asynccnt 0x0" ::: "memory");
}

// ---------------------------------------------------------------- cast f32->bf16
__global__ void cast_f32_bf16(const float* __restrict__ in, __bf16* __restrict__ out, int n) {
  int i = blockIdx.x * blockDim.x + threadIdx.x;
  if (i < n) out[i] = (__bf16)in[i];
}

// ---------------------------------------------------------------- generic bf16 WMMA GEMM
// C(M,N) = A(M,K) * W(N,K)^T + bias(N)
// block = 256 threads = 8 waves (2 M-tiles x 4 N-tiles of 16x64) -> 32x256 block tile
// Software-pipelined: ping-pong fragment sets so loads of chunk k+32 overlap WMMAs of k.
// mode 0: out[row*ldo + coloff + n]
// mode 1: head-split-64: out[row*ldo + (n>>6)*128 + coloff + (n&63)]
// mode 2: V-transpose:   out[((b*HD + n>>7)*DH + (n&127))*SS + (row&2047)]
__global__ void gemm_bf16_wmma(const __bf16* __restrict__ A, const __bf16* __restrict__ W,
                               const float* __restrict__ bias, int K,
                               float* __restrict__ outF, __bf16* __restrict__ outB,
                               int ldo, int mode, int coloff) {
  const int lane = threadIdx.x & 31;
  const int wave = threadIdx.x >> 5;
  const int half = lane >> 4;
  const int ln   = lane & 15;
  const int mrow = blockIdx.y * 32 + (wave >> 2) * 16;
  const int nb0  = blockIdx.x * 256 + (wave & 3) * 64;

  FragC c[4];
#pragma unroll
  for (int t = 0; t < 4; ++t)
#pragma unroll
    for (int r = 0; r < 8; ++r) c[t].f[r] = 0.0f;

  const __bf16* arow = A + (size_t)(mrow + ln) * K;
  const __bf16* wrow[4];
#pragma unroll
  for (int t = 0; t < 4; ++t)
    wrow[t] = W + (size_t)(nb0 + t * 16 + ln) * K;

  FragAB aA, aB, bA[4], bB[4];

  auto ldA = [&](FragAB& a, int kb) {
#pragma unroll
    for (int g = 0; g < 8; ++g)
      a.u[g] = *(const unsigned int*)(arow + kb + koffA(g, half));
  };
  auto ldB = [&](FragAB* bb, int kb) {
#pragma unroll
    for (int t = 0; t < 4; ++t)
#pragma unroll
      for (int g = 0; g < 8; ++g)
        bb[t].u[g] = *(const unsigned int*)(wrow[t] + kb + koffA(g, half));
  };

  ldA(aA, 0);
  ldB(bA, 0);
  for (int kb = 0; kb < K; kb += 64) {       // K is always a multiple of 64
    __builtin_prefetch(arow + kb + 256, 0, 0);   // global_prefetch_b8
    ldA(aB, kb + 32);
    ldB(bB, kb + 32);
#pragma unroll
    for (int t = 0; t < 4; ++t)
      c[t].v = __builtin_amdgcn_wmma_f32_16x16x32_bf16(
          false, aA.v, false, bA[t].v, (short)0, c[t].v, false, false);
    int kn = (kb + 64 < K) ? (kb + 64) : 0;  // tail: benign redundant reload
    ldA(aA, kn);
    ldB(bA, kn);
#pragma unroll
    for (int t = 0; t < 4; ++t)
      c[t].v = __builtin_amdgcn_wmma_f32_16x16x32_bf16(
          false, aB.v, false, bB[t].v, (short)0, c[t].v, false, false);
  }

#pragma unroll
  for (int t = 0; t < 4; ++t) {
    int ncol = nb0 + t * 16 + ln;
    float bv = bias[ncol];
#pragma unroll
    for (int r = 0; r < 8; ++r) {
      int row = mrow + r + half * 8;
      float val = c[t].f[r] + bv;
      size_t idx;
      if (mode == 0) {
        idx = (size_t)row * ldo + coloff + ncol;
      } else if (mode == 1) {
        idx = (size_t)row * ldo + (ncol >> 6) * 128 + coloff + (ncol & 63);
      } else {
        int bidx = row >> 11, s = row & (SS - 1);
        int h = ncol >> 7, d = ncol & (DH - 1);
        idx = (size_t)(((bidx * HD + h) * DH + d)) * SS + s;
      }
      if (outF) outF[idx] = val;
      if (outB) outB[idx] = (__bf16)val;
    }
  }
}

// ---------------------------------------------------------------- RoPE + scatter into Q/K
// raw: TT x 1024 (t, h*64 + r) f32 -> dst col (h*128 + 64 + r) bf16
__global__ void rope_write(const float* __restrict__ raw, __bf16* __restrict__ dst) {
  int i = blockIdx.x * blockDim.x + threadIdx.x;
  if (i >= TT * 1024) return;
  int t = i >> 10;
  int cidx = i & 1023;
  int h = cidx >> 6, r = cidx & 63;
  int s = t & (SS - 1);
  int f = r & 31;
  float inv = __expf(-__logf(10000.0f) * (float)(2 * f) * (1.0f / 64.0f));
  float sn, cs;
  __sincosf((float)s * inv, &sn, &cs);
  float xv = raw[i];
  int pr = (r < 32) ? r + 32 : r - 32;
  float partner = raw[(t << 10) + (h << 6) + pr];
  float rot = (r < 32) ? -partner : partner;
  dst[(size_t)t * EE + h * DH + 64 + r] = (__bf16)(xv * cs + rot * sn);
}

// ---------------------------------------------------------------- flash attention (WMMA)
// Q,K: (b,s,h,d) at ((b*SS+s)*EE + h*DH + d) bf16 ; Vt: ((b*HD+h)*DH+d)*SS + s bf16
// Block = 8 waves, one (b,h) and 128 query rows. Per 32-key chunk, the block stages
// K (32x128) and V^T (128x32) tiles into LDS with async-to-LDS DMA, double buffered:
// chunk j+1's DMA overlaps chunk j's WMMAs/softmax.
__global__ void mla_attention(const __bf16* __restrict__ Q, const __bf16* __restrict__ Kb,
                              const __bf16* __restrict__ Vt, __bf16* __restrict__ AO) {
  __shared__ __align__(16) __bf16 Kt[2][32 * 128];    // [key][d]
  __shared__ __align__(16) __bf16 Vtile[2][128 * 32]; // [d][key]
  __shared__ __align__(16) __bf16 Pl[8 * 512];        // per-wave 16x32 P tile
  const int tid  = threadIdx.x;
  const int lane = tid & 31;
  const int wave = tid >> 5;
  const int half = lane >> 4;
  const int ln   = lane & 15;
  const int bh   = blockIdx.x >> 4;          // (b*HD + h)
  const int qblk = blockIdx.x & 15;
  const int b = bh >> 4, h = bh & 15;
  const int qbase = qblk * 128 + wave * 16;
  __bf16* Pw = Pl + wave * 512;

  const float scale = 0.08838834764831845f;  // 1/sqrt(128)

  // preload Q fragments (registers, reused for all key chunks)
  FragAB qf[4];
#pragma unroll
  for (int kc = 0; kc < 4; ++kc) {
    const __bf16* qp = Q + (size_t)(b * SS + qbase + ln) * EE + h * DH + kc * 32;
#pragma unroll
    for (int g = 0; g < 8; ++g)
      qf[kc].u[g] = *(const unsigned int*)(qp + koffA(g, half));
  }

  // issue one chunk's K+V tiles as async-to-LDS DMA (whole block cooperates)
  auto stage = [&](int jb, int buf) {
#pragma unroll
    for (int i = 0; i < 2; ++i) {
      int t16 = tid * 2 + i;                       // 512 16B segments each
      int kr = t16 >> 4, kseg = t16 & 15;          // K: 32 rows x 256B
      async_copy16(Kb + (size_t)(b * SS + jb + kr) * EE + h * DH + kseg * 8,
                   &Kt[buf][kr * 128 + kseg * 8]);
      int vd = t16 >> 2, vseg = t16 & 3;           // V^T: 128 rows x 64B
      async_copy16(Vt + (size_t)((b * HD + h) * DH + vd) * SS + jb + vseg * 8,
                   &Vtile[buf][vd * 32 + vseg * 8]);
    }
  };

  float mrow[8], lrow[8];
  FragC o[8];
#pragma unroll
  for (int r = 0; r < 8; ++r) { mrow[r] = -1e30f; lrow[r] = 0.0f; }
#pragma unroll
  for (int t = 0; t < 8; ++t)
#pragma unroll
    for (int r = 0; r < 8; ++r) o[t].f[r] = 0.0f;

  stage(0, 0);
  for (int j = 0; j < SS / 32; ++j) {
    wait_async0();        // this wave's DMA for chunk j landed
    __syncthreads();      // everyone's landed; chunk j-1's consumers all done
    if (j + 1 < SS / 32) stage((j + 1) * 32, (j + 1) & 1);
    const __bf16* Kc = Kt[j & 1];
    const __bf16* Vc = Vtile[j & 1];

    // S(16x32) = Q(16x128) x K_chunk^T, from LDS
    FragC s0, s1;
#pragma unroll
    for (int r = 0; r < 8; ++r) { s0.f[r] = 0.0f; s1.f[r] = 0.0f; }
#pragma unroll
    for (int kc = 0; kc < 4; ++kc) {
      FragAB b0, b1;
#pragma unroll
      for (int g = 0; g < 8; ++g) {
        b0.u[g] = *(const unsigned int*)(Kc + ln * 128        + kc * 32 + koffA(g, half));
        b1.u[g] = *(const unsigned int*)(Kc + (16 + ln) * 128 + kc * 32 + koffA(g, half));
      }
      s0.v = __builtin_amdgcn_wmma_f32_16x16x32_bf16(false, qf[kc].v, false, b0.v, (short)0, s0.v, false, false);
      s1.v = __builtin_amdgcn_wmma_f32_16x16x32_bf16(false, qf[kc].v, false, b1.v, (short)0, s1.v, false, false);
    }

    // online softmax (row = r + 8*half lives across the 16 lanes of this half)
    float p0[8], p1[8];
#pragma unroll
    for (int r = 0; r < 8; ++r) {
      float v0 = s0.f[r] * scale, v1 = s1.f[r] * scale;
      float mloc = fmaxf(v0, v1);
      for (int msk = 1; msk < 16; msk <<= 1) mloc = fmaxf(mloc, __shfl_xor(mloc, msk, 32));
      float mnew = fmaxf(mrow[r], mloc);
      float alpha = __expf(mrow[r] - mnew);
      mrow[r] = mnew;
      float e0 = __expf(v0 - mnew), e1 = __expf(v1 - mnew);
      float ps = e0 + e1;
      for (int msk = 1; msk < 16; msk <<= 1) ps += __shfl_xor(ps, msk, 32);
      lrow[r] = lrow[r] * alpha + ps;
      p0[r] = e0; p1[r] = e1;
#pragma unroll
      for (int t = 0; t < 8; ++t) o[t].f[r] *= alpha;
    }

    // C-layout -> A-layout via per-wave LDS tile (same-wave DS ops stay in order)
#pragma unroll
    for (int r = 0; r < 8; ++r) {
      int rw = r + half * 8;
      Pw[rw * 32 + ln]      = (__bf16)p0[r];
      Pw[rw * 32 + 16 + ln] = (__bf16)p1[r];
    }
    __builtin_amdgcn_wave_barrier();
    FragAB pf;
#pragma unroll
    for (int g = 0; g < 8; ++g)
      pf.u[g] = *(const unsigned int*)(Pw + ln * 32 + koffA(g, half));
    __builtin_amdgcn_wave_barrier();

    // O(16x128) += P(16x32) x V(32x128), V B-fragments from LDS
#pragma unroll
    for (int t = 0; t < 8; ++t) {
      FragAB vf;
#pragma unroll
      for (int g = 0; g < 8; ++g)
        vf.u[g] = *(const unsigned int*)(Vc + (t * 16 + ln) * 32 + koffA(g, half));
      o[t].v = __builtin_amdgcn_wmma_f32_16x16x32_bf16(false, pf.v, false, vf.v, (short)0, o[t].v, false, false);
    }
  }

#pragma unroll
  for (int r = 0; r < 8; ++r) {
    float rl = 1.0f / lrow[r];
    int row = qbase + r + half * 8;
#pragma unroll
    for (int t = 0; t < 8; ++t)
      AO[(size_t)(b * SS + row) * EE + h * DH + t * 16 + ln] = (__bf16)(o[t].f[r] * rl);
  }
}

// ---------------------------------------------------------------- launch
extern "C" void kernel_launch(void* const* d_in, const int* in_sizes, int n_in,
                              void* d_out, int out_size, void* d_ws, size_t ws_size,
                              hipStream_t stream) {
  (void)in_sizes; (void)n_in; (void)out_size; (void)ws_size;
  const float* x     = (const float*)d_in[0];
  const float* Wkv_d = (const float*)d_in[1];
  const float* bkv_d = (const float*)d_in[2];
  const float* Wq_d  = (const float*)d_in[3];
  const float* bq_d  = (const float*)d_in[4];
  const float* Wk_u  = (const float*)d_in[5];
  const float* bk_u  = (const float*)d_in[6];
  const float* Wq_u  = (const float*)d_in[7];
  const float* bq_u  = (const float*)d_in[8];
  const float* Wv_u  = (const float*)d_in[9];
  const float* bv_u  = (const float*)d_in[10];
  const float* Wrk   = (const float*)d_in[11];
  const float* brk   = (const float*)d_in[12];
  const float* Wrq   = (const float*)d_in[13];
  const float* brq   = (const float*)d_in[14];
  const float* Wo    = (const float*)d_in[15];
  const float* bo    = (const float*)d_in[16];
  float* out = (float*)d_out;

  char* ws = (char*)d_ws;
  size_t off = 0;
  auto alloc = [&](size_t bytes) -> void* {
    void* p = ws + off;
    off += (bytes + 255) & ~(size_t)255;
    return p;
  };

  __bf16* xb   = (__bf16*)alloc((size_t)TT * EE * 2);
  __bf16* wWkv = (__bf16*)alloc((size_t)LATD * EE * 2);
  __bf16* wWqd = (__bf16*)alloc((size_t)LATD * EE * 2);
  __bf16* wWku = (__bf16*)alloc((size_t)1024 * LATD * 2);
  __bf16* wWqu = (__bf16*)alloc((size_t)1024 * LATD * 2);
  __bf16* wWvu = (__bf16*)alloc((size_t)EE * LATD * 2);
  __bf16* wWrk = (__bf16*)alloc((size_t)1024 * EE * 2);
  __bf16* wWrq = (__bf16*)alloc((size_t)1024 * LATD * 2);
  __bf16* wWo  = (__bf16*)alloc((size_t)EE * EE * 2);
  __bf16* kvd  = (__bf16*)alloc((size_t)TT * LATD * 2);
  __bf16* qd   = (__bf16*)alloc((size_t)TT * LATD * 2);
  __bf16* Kbuf = (__bf16*)alloc((size_t)TT * EE * 2);
  __bf16* Qbuf = (__bf16*)alloc((size_t)TT * EE * 2);
  __bf16* Vt   = (__bf16*)alloc((size_t)TT * EE * 2);
  float*  raw  = (float*)alloc((size_t)TT * 1024 * 4);
  __bf16* AO   = xb;  // xb dead after the Wrk GEMM; reuse for attention output

  auto cast = [&](const float* src, __bf16* dst, size_t n) {
    cast_f32_bf16<<<dim3((unsigned)((n + 255) / 256)), dim3(256), 0, stream>>>(src, dst, (int)n);
  };
  cast(x, xb, (size_t)TT * EE);
  cast(Wkv_d, wWkv, (size_t)LATD * EE);
  cast(Wq_d,  wWqd, (size_t)LATD * EE);
  cast(Wk_u,  wWku, (size_t)1024 * LATD);
  cast(Wq_u,  wWqu, (size_t)1024 * LATD);
  cast(Wv_u,  wWvu, (size_t)EE * LATD);
  cast(Wrk,   wWrk, (size_t)1024 * EE);
  cast(Wrq,   wWrq, (size_t)1024 * LATD);
  cast(Wo,    wWo,  (size_t)EE * EE);

  auto gemm = [&](const __bf16* A, const __bf16* W, const float* bias, int M, int N, int K,
                  float* oF, __bf16* oB, int ldo, int mode, int coloff) {
    dim3 grid(N / 256, M / 32);
    gemm_bf16_wmma<<<grid, dim3(256), 0, stream>>>(A, W, bias, K, oF, oB, ldo, mode, coloff);
  };

  gemm(xb,  wWkv, bkv_d, TT, 512,  2048, nullptr, kvd,  512,  0, 0);  // kv_d
  gemm(xb,  wWqd, bq_d,  TT, 512,  2048, nullptr, qd,   512,  0, 0);  // q_d
  gemm(kvd, wWku, bk_u,  TT, 1024, 512,  nullptr, Kbuf, 2048, 1, 0);  // k1 -> K[:, h*128 + 0..63]
  gemm(qd,  wWqu, bq_u,  TT, 1024, 512,  nullptr, Qbuf, 2048, 1, 0);  // q1 -> Q[:, h*128 + 0..63]
  gemm(kvd, wWvu, bv_u,  TT, 2048, 512,  nullptr, Vt,   0,    2, 0);  // v  -> V^T (b,h,d,s)
  gemm(xb,  wWrk, brk,   TT, 1024, 2048, raw,     nullptr, 1024, 0, 0); // kr raw
  rope_write<<<dim3((TT * 1024) / 256), dim3(256), 0, stream>>>(raw, Kbuf);
  gemm(qd,  wWrq, brq,   TT, 1024, 512,  raw,     nullptr, 1024, 0, 0); // qr raw
  rope_write<<<dim3((TT * 1024) / 256), dim3(256), 0, stream>>>(raw, Qbuf);

  mla_attention<<<dim3(512), dim3(256), 0, stream>>>(Qbuf, Kbuf, Vt, AO);

  gemm(AO, wWo, bo, TT, 2048, 2048, out, nullptr, 2048, 0, 0);        // output proj
}
